// ConvAttnPool_old_85899346173
// MI455X (gfx1250) — matrix operand order repaired and checked by reading, not compiled, not run on hardware
//
#include <hip/hip_runtime.h>
#include <hip/hip_bf16.h>
#include <stdint.h>

typedef __attribute__((ext_vector_type(16))) __bf16 v16bf;
typedef __attribute__((ext_vector_type(8)))  float  v8f;

// Problem constants
constexpr int cB  = 4;
constexpr int cL  = 2500;
constexpr int cLP = 2560;   // L padded (zero-padded h) so WMMA tiles need no edge guards
constexpr int cY  = 8922;
constexpr int cE  = 100;
constexpr int cF  = 50;
constexpr int cFP = 64;     // F padded for WMMA K/N dims (zero-padded)
constexpr int cK  = 9;

// ---------------------------------------------------------------------------
// Kernel 1: embedding gather -> conv1d (same) -> tanh.
// The embedding tile (+halo) is gathered with GLOBAL_LOAD_ASYNC_TO_LDS_B128:
// per-lane global address -> per-lane LDS address, no VGPR round trip,
// tracked with ASYNCcnt. Note embed_table[0] == 0 in setup, so gathering
// token 0 already yields the zero row demanded by padding_idx=0.
// Writes h in bf16 twice: h[b][l][fp] (f-contiguous, B-operand of scores)
// and hT[b][fp][lp] (l-contiguous, B-operand of pooling).
// ---------------------------------------------------------------------------
__global__ void k_embed_conv(const int* __restrict__ x,
                             const float* __restrict__ table,
                             const float* __restrict__ cw,
                             const float* __restrict__ cb,
                             __bf16* __restrict__ h,
                             __bf16* __restrict__ hT) {
  const int ltile = blockIdx.x;           // 40 tiles of 64 positions
  const int b     = blockIdx.y;
  const int lbase = ltile * 64;
  const int tid   = threadIdx.x;

  __shared__ int tok[72];
  __shared__ __align__(16) float emb[72][cE];   // 64 + 8 halo rows, ~28.8 KB

  for (int i = tid; i < 72; i += 256) {
    int l = lbase + i - (cK / 2);
    tok[i] = (l >= 0 && l < cL) ? x[b * cL + l] : 0;
  }
  __syncthreads();

  // Async DMA gather: 72 rows x 25 chunks of 16B (E=100 floats = 400B/row).
  for (int i = tid; i < 72 * 25; i += 256) {
    int r = i / 25, c = i % 25;
    const float* gp = table + (size_t)tok[r] * cE + c * 4;
    uint32_t ldsoff = (uint32_t)(uintptr_t)(&emb[r][c * 4]);
    uint64_t gaddr  = (uint64_t)(uintptr_t)gp;
    asm volatile("global_load_async_to_lds_b128 %0, %1, off"
                 :: "v"(ldsoff), "v"(gaddr) : "memory");
  }
  asm volatile("s_wait_asynccnt 0x0" ::: "memory");
  __syncthreads();

  for (int o = tid; o < 64 * cFP; o += 256) {
    int lo = o / cFP;
    int f  = o % cFP;
    int l  = lbase + lo;
    float acc = 0.0f;
    if (f < cF && l < cL) {
      acc = cb[f];
      const float* wrow = cw + (size_t)f * cE * cK;     // [E][K]
      for (int e = 0; e < cE; ++e) {
        const float* wk = wrow + e * cK;
        #pragma unroll
        for (int k = 0; k < cK; ++k)
          acc += emb[lo + k][e] * wk[k];
      }
      acc = tanhf(acc);
    }
    __bf16 hv = (__bf16)acc;
    h [((size_t)b * cLP + l) * cFP + f] = hv;
    hT[((size_t)b * cFP + f) * cLP + l] = hv;
  }
}

// ---------------------------------------------------------------------------
// Kernel 2: scores[b,y,l] = sum_f U[y,f] * h[b,l,f]   (bf16 WMMA, f32 acc)
// A (16x32, K=f): lane n -> M=n%16; elems[0..15]=K 16*(n/16)..16*(n/16)+15.
// B (32x16, K=f, N=l): lane n -> col lbase+n%16, 16 contiguous K -> one 32B load.
// D: VGPR r -> scores[ybase + r + 8*(n/16)][lbase + n%16].
// Software-pipelined: B operands for tile t+4 are loaded before the WMMAs of
// tile t so global loads overlap compute + NT score stores.
// ---------------------------------------------------------------------------
__global__ void k_scores(const float* __restrict__ Uw,
                         const __bf16* __restrict__ h,
                         float* __restrict__ alpha) {
  const int b     = blockIdx.y;
  const int ybase = blockIdx.x * 16;
  const int lane  = threadIdx.x & 31;
  const int wv    = threadIdx.x >> 5;     // 4 waves split l-tiles
  const int m     = lane & 15;
  const int half  = lane >> 4;
  const int base8 = half * 8;

  const int  yy  = ybase + m;
  const bool yok = (yy < cY);
  const bool ytileFull = (ybase + 16 <= cY);   // wave-uniform
  const float* ur = Uw + (size_t)(yok ? yy : 0) * cF;

  v16bf a0, a1;
  #pragma unroll
  for (int j = 0; j < 8; ++j) {
    int f0 = 2 * base8 + j, f1 = 8 + 2 * base8 + j, f2 = 32 + 2 * base8 + j, f3 = 40 + 2 * base8 + j;
    a0[j]     = (__bf16)(yok ? ur[f0] : 0.0f);              // f0 < 32 < F
    a0[8 + j] = (__bf16)(yok ? ur[f1] : 0.0f);              // f1 < 32 < F
    a1[j]     = (__bf16)((yok && f2 < cF) ? ur[f2] : 0.0f);
    a1[8 + j] = (__bf16)((yok && f3 < cF) ? ur[f3] : 0.0f);
  }

  const __bf16* hbase = h + (size_t)b * cLP * cFP + half * 16;
  float* arow = alpha + (size_t)b * cY * cL;

  const int ntiles = cLP / 16;            // 160
  v16bf b0, b1;
  {
    const __bf16* hp = hbase + (size_t)(wv * 16 + m) * cFP;
    b0 = *(const v16bf*)hp;
    b1 = *(const v16bf*)(hp + 32);
  }
  v16bf n0 = b0, n1 = b1;

  for (int t = wv; t < ntiles; t += 4) {
    // Prefetch + preload next tile's B operands (pipeline stage t+4)
    const int tn = t + 4;
    if (tn < ntiles) {
      const __bf16* hp = hbase + (size_t)(tn * 16 + m) * cFP;
      __builtin_prefetch(hp + (size_t)4 * 16 * cFP, 0, 1);
      n0 = *(const v16bf*)hp;
      n1 = *(const v16bf*)(hp + 32);
    }

    v8f c = {0.f, 0.f, 0.f, 0.f, 0.f, 0.f, 0.f, 0.f};
    c = __builtin_amdgcn_wmma_f32_16x16x32_bf16(false, a0, false, b0, (short)0, c, false, false);
    c = __builtin_amdgcn_wmma_f32_16x16x32_bf16(false, a1, false, b1, (short)0, c, false, false);

    const int lcol = t * 16 + m;
    if (lcol < cL) {
      float* sp = arow + (size_t)(ybase + half * 8) * cL + lcol;
      if (ytileFull) {            // uniform fast path: 8 unconditional NT stores
        #pragma unroll
        for (int r = 0; r < 8; ++r)
          __builtin_nontemporal_store(c[r], sp + (size_t)r * cL);
      } else {
        #pragma unroll
        for (int r = 0; r < 8; ++r)
          if (ybase + half * 8 + r < cY)
            __builtin_nontemporal_store(c[r], sp + (size_t)r * cL);
      }
    }
    b0 = n0; b1 = n1;
  }
}

// ---------------------------------------------------------------------------
// Kernel 3: per-(b,y) online softmax stats over L: rowmax, 1/sum(exp).
// One block of 128 threads per row; single NT streaming read pass.
// ---------------------------------------------------------------------------
__global__ void k_softmax_stats(const float* __restrict__ alpha,
                                float* __restrict__ rowmax,
                                float* __restrict__ rowinv) {
  const int row = blockIdx.x;             // b*Y + y
  const float* s = alpha + (size_t)row * cL;

  float mx = -3.0e38f, sum = 0.0f;
  for (int l = threadIdx.x; l < cL; l += blockDim.x) {
    float v = __builtin_nontemporal_load(&s[l]);
    if (v > mx) { sum = sum * __expf(mx - v) + 1.0f; mx = v; }
    else        { sum += __expf(v - mx); }
  }
  #pragma unroll
  for (int off = 16; off > 0; off >>= 1) {
    float mo = __shfl_xor(mx,  off, 32);
    float so = __shfl_xor(sum, off, 32);
    float nm = fmaxf(mx, mo);
    sum = sum * __expf(mx - nm) + so * __expf(mo - nm);
    mx = nm;
  }
  __shared__ float sm[4], ss[4];
  const int lane = threadIdx.x & 31, wvi = threadIdx.x >> 5;
  if (lane == 0) { sm[wvi] = mx; ss[wvi] = sum; }
  __syncthreads();
  if (threadIdx.x == 0) {
    float M = sm[0], S = ss[0];
    for (int w = 1; w < 4; ++w) {
      float nm = fmaxf(M, sm[w]);
      S = S * __expf(M - nm) + ss[w] * __expf(sm[w] - nm);
      M = nm;
    }
    rowmax[row] = M;
    rowinv[row] = 1.0f / S;
  }
}

// ---------------------------------------------------------------------------
// Kernel 4 (fused): alpha = exp(s-max)*inv written back (NT), and
// m[b,y,f] = sum_l p * h[b,l,f] via bf16 WMMA (A: K=l chunk of 32; B from hT
// so each lane pulls 16 contiguous l for its column f). One wave per y-tile.
// Uniform fast path for full chunks (kbase+32 <= L).
// ---------------------------------------------------------------------------
__global__ void k_attn_pool(const __bf16* __restrict__ hT,
                            const float* __restrict__ rowmax,
                            const float* __restrict__ rowinv,
                            float* __restrict__ alpha,
                            float* __restrict__ mout) {
  const int b     = blockIdx.y;
  const int ybase = blockIdx.x * 16;
  const int lane  = threadIdx.x & 31;
  const int mm    = lane & 15;
  const int half  = lane >> 4;
  const int base8 = half * 8;

  const int  y   = ybase + mm;
  const bool yok = (y < cY);
  const bool ytileFull = (ybase + 16 <= cY);     // wave-uniform
  const float rmax = yok ? rowmax[b * cY + y] : 0.0f;
  const float rinv = yok ? rowinv[b * cY + y] : 0.0f;
  float* srow = alpha + ((size_t)(b * cY + (yok ? y : 0))) * cL;

  v8f c0 = {0.f,0.f,0.f,0.f,0.f,0.f,0.f,0.f};
  v8f c1 = c0, c2 = c0, c3 = c0;

  const __bf16* hTb = hT + (size_t)b * cFP * cLP;

  for (int chunk = 0; chunk < cLP / 32; ++chunk) {  // 80 chunks
    const int kbase = chunk * 32;
    const bool chunkFull = (kbase + 32 <= cL);      // wave-uniform

    v16bf a;
    if (ytileFull && chunkFull) {
      #pragma unroll
      for (int j = 0; j < 8; ++j) {
        int l0 = kbase + 2 * base8 + j;
        int l1 = kbase + 8 + 2 * base8 + j;
        float p0 = __expf(srow[l0] - rmax);
        float p1 = __expf(srow[l1] - rmax);
        a[j]     = (__bf16)p0;
        a[8 + j] = (__bf16)p1;
        __builtin_nontemporal_store(p0 * rinv, &srow[l0]);
        __builtin_nontemporal_store(p1 * rinv, &srow[l1]);
      }
    } else {
      #pragma unroll
      for (int j = 0; j < 8; ++j) {
        int l0 = kbase + 2 * base8 + j;
        int l1 = kbase + 8 + 2 * base8 + j;
        float p0 = (yok && l0 < cL) ? __expf(srow[l0] - rmax) : 0.0f;
        float p1 = (yok && l1 < cL) ? __expf(srow[l1] - rmax) : 0.0f;
        a[j]     = (__bf16)p0;
        a[8 + j] = (__bf16)p1;
        if (yok && l0 < cL) __builtin_nontemporal_store(p0 * rinv, &srow[l0]);
        if (yok && l1 < cL) __builtin_nontemporal_store(p1 * rinv, &srow[l1]);
      }
    }

    const __bf16* hp = hTb + kbase + half * 16;
    v16bf b0 = *(const v16bf*)(hp + (size_t)(mm)      * cLP);
    v16bf b1 = *(const v16bf*)(hp + (size_t)(16 + mm) * cLP);
    v16bf b2 = *(const v16bf*)(hp + (size_t)(32 + mm) * cLP);
    v16bf b3 = *(const v16bf*)(hp + (size_t)(48 + mm) * cLP);

    c0 = __builtin_amdgcn_wmma_f32_16x16x32_bf16(false, a, false, b0, (short)0, c0, false, false);
    c1 = __builtin_amdgcn_wmma_f32_16x16x32_bf16(false, a, false, b1, (short)0, c1, false, false);
    c2 = __builtin_amdgcn_wmma_f32_16x16x32_bf16(false, a, false, b2, (short)0, c2, false, false);
    c3 = __builtin_amdgcn_wmma_f32_16x16x32_bf16(false, a, false, b3, (short)0, c3, false, false);
  }

  float* mbase = mout + ((size_t)(b * cY + ybase + half * 8)) * cFP + mm;
  if (ytileFull) {
    #pragma unroll
    for (int r = 0; r < 8; ++r) {
      float* mr = mbase + (size_t)r * cFP;
      mr[0]  = c0[r];
      mr[16] = c1[r];
      mr[32] = c2[r];
      mr[48] = c3[r];
    }
  } else {
    #pragma unroll
    for (int r = 0; r < 8; ++r) {
      if (ybase + half * 8 + r < cY) {
        float* mr = mbase + (size_t)r * cFP;
        mr[0]  = c0[r];
        mr[16] = c1[r];
        mr[32] = c2[r];
        mr[48] = c3[r];
      }
    }
  }
}

// ---------------------------------------------------------------------------
// Kernel 5: y = final_w . m + final_b ; yhat = sigmoid(y); per-row BCE term
// ---------------------------------------------------------------------------
__global__ void k_output(const float* __restrict__ mbuf,
                         const float* __restrict__ fw,
                         const float* __restrict__ fb,
                         const float* __restrict__ target,
                         float* __restrict__ yhat,
                         float* __restrict__ lterms) {
  int i = blockIdx.x * blockDim.x + threadIdx.x;
  if (i >= cB * cY) return;
  int y = i % cY;
  const float* mr = mbuf + (size_t)i * cFP;
  const float* wr = fw + (size_t)y * cF;
  float acc = fb[y];
  for (int f = 0; f < cF; ++f) acc += wr[f] * mr[f];
  yhat[i] = 1.0f / (1.0f + __expf(-acc));
  float t = target[i];
  lterms[i] = fmaxf(acc, 0.0f) - acc * t + log1pf(__expf(-fabsf(acc)));
}

// ---------------------------------------------------------------------------
// Kernel 6: deterministic mean reduction of BCE terms -> loss scalar
// ---------------------------------------------------------------------------
__global__ void k_loss_reduce(const float* __restrict__ terms,
                              float* __restrict__ loss) {
  __shared__ float red[256];
  float acc = 0.0f;
  for (int i = threadIdx.x; i < cB * cY; i += 256) acc += terms[i];
  red[threadIdx.x] = acc;
  __syncthreads();
  for (int s = 128; s > 0; s >>= 1) {
    if (threadIdx.x < s) red[threadIdx.x] += red[threadIdx.x + s];
    __syncthreads();
  }
  if (threadIdx.x == 0) *loss = red[0] / (float)(cB * cY);
}

// ---------------------------------------------------------------------------
extern "C" void kernel_launch(void* const* d_in, const int* in_sizes, int n_in,
                              void* d_out, int out_size, void* d_ws, size_t ws_size,
                              hipStream_t stream) {
  const int*   x      = (const int*)  d_in[0];
  const float* target = (const float*)d_in[1];
  const float* table  = (const float*)d_in[2];
  const float* cw     = (const float*)d_in[3];
  const float* cb     = (const float*)d_in[4];
  const float* Uw     = (const float*)d_in[5];
  const float* fw     = (const float*)d_in[6];
  const float* fb     = (const float*)d_in[7];

  float* out   = (float*)d_out;
  float* yhat  = out;                       // [B*Y]
  float* loss  = out + cB * cY;             // [1]
  float* alpha = out + cB * cY + 1;         // [B*Y*L] (raw scores -> alpha)

  // Workspace carve-up (256B aligned)
  char* ws = (char*)d_ws;
  size_t off = 0;
  auto carve = [&](size_t bytes) {
    size_t o = off;
    off = (off + bytes + 255) & ~(size_t)255;
    return o;
  };
  __bf16* h      = (__bf16*)(ws + carve((size_t)cB * cLP * cFP * 2));
  __bf16* hT     = (__bf16*)(ws + carve((size_t)cB * cFP * cLP * 2));
  float*  rowmax = (float*) (ws + carve((size_t)cB * cY * 4));
  float*  rowinv = (float*) (ws + carve((size_t)cB * cY * 4));
  float*  mbuf   = (float*) (ws + carve((size_t)cB * cY * cFP * 4));
  float*  lterms = (float*) (ws + carve((size_t)cB * cY * 4));

  const int ytiles = (cY + 15) / 16;        // 558

  k_embed_conv   <<<dim3(cLP / 64, cB), 256, 0, stream>>>(x, table, cw, cb, h, hT);
  k_scores       <<<dim3(ytiles, cB),   128, 0, stream>>>(Uw, h, alpha);
  k_softmax_stats<<<cB * cY,            128, 0, stream>>>(alpha, rowmax, rowinv);
  k_attn_pool    <<<dim3(ytiles, cB),    32, 0, stream>>>(hT, rowmax, rowinv, alpha, mbuf);
  k_output       <<<(cB * cY + 255)/256, 256, 0, stream>>>(mbuf, fw, fb, target, yhat, lterms);
  k_loss_reduce  <<<1,                  256, 0, stream>>>(lterms, loss);
}
